// Model_22016002360008
// MI455X (gfx1250) — matrix operand order repaired and checked by reading, not compiled, not run on hardware
//
#include <hip/hip_runtime.h>
#include <math.h>

// ---------------- constants ----------------
#define N_ATOMS 64
#define NS      32768          // N_SAMPLES
#define KDIM    4096           // N_RES
#define TOTAL   (N_ATOMS * NS) // 2097152
#define MCONV   32768          // complex FFT size for 65536-pt real conv (LDS 256KB)
#define LOG_MCONV 15
#define MFILT   16384          // complex FFT size for 32768-pt real filter (LDS 128KB)
#define LOG_MFILT 14
#define NCOEF   16385
#define PI_F    3.14159265358979f

// ---------------- WMMA types ----------------
typedef __attribute__((ext_vector_type(16))) __bf16 v16bf;
typedef __attribute__((ext_vector_type(8)))  float  v8f;

// ---------------- gfx1250 async-to-LDS helpers (ASYNCcnt path) -------------
__device__ inline void async_load_b128(unsigned lds_off, const void* gptr) {
  asm volatile("global_load_async_to_lds_b128 %0, %1, off"
               :: "v"(lds_off), "v"((unsigned long long)(size_t)gptr)
               : "memory");
}
__device__ inline void wait_asynccnt0() {
  asm volatile("s_wait_asynccnt 0" ::: "memory");
}

// ---------------- per-atom params ----------------
struct AtomParams {
  float m0[N_ATOMS], m1[N_ATOMS];
  float mu_n[N_ATOMS], sg_n[N_ATOMS];
  float sg1[N_ATOMS], sg2[N_ATOMS];
  float mu_e[N_ATOMS], sg_e[N_ATOMS];
  float amp[N_ATOMS];
  float mixv[N_ATOMS];
  int   room[N_ATOMS];
  float ld_d[N_ATOMS], ld_f[N_ATOMS];
};

// ---------------- small helpers ----------------
__device__ inline float sigmoidf_(float x) { return 1.f / (1.f + expf(-x)); }
__device__ inline float leakyf_(float x)   { return x > 0.f ? x : 0.01f * x; }

__device__ inline unsigned f2bfraw(float f) {      // rounded bf16 in bits [31:16]
  unsigned u = __builtin_bit_cast(unsigned, f);
  return (u + 0x7FFFu + ((u >> 16) & 1u)) & 0xFFFF0000u;
}
__device__ inline unsigned short f2bfbits(float f) {
  return (unsigned short)(f2bfraw(f) >> 16);
}
// pack two floats into one dword of bf16s: low half = a, high half = b
__device__ inline unsigned pack2bf(float a, float b) {
  return (f2bfraw(a) >> 16) | f2bfraw(b);
}

// exponential_decay with broadcast decay value == d^(frame+1), linearly
// interpolated from 128 frames to NS samples.
__device__ inline float decay_interp(float ld, int i) {
  float pos = ((float)i + 0.5f) * (128.f / (float)NS) - 0.5f;
  pos = fminf(fmaxf(pos, 0.f), 127.f);
  float fi = floorf(pos);
  int i0 = (int)fi;
  int i1 = min(i0 + 1, 127);
  float w = pos - fi;
  return expf(ld * (float)(i0 + 1)) * (1.f - w) + expf(ld * (float)(i1 + 1)) * w;
}

__device__ inline float gaussf_(float x, float mu, float sg) {
  float t = (x - mu) / sg;
  return expf(-0.5f * t * t) / (sg * 2.5066282746310002f);
}

__device__ inline float2 cmul_(float2 a, float2 b) {
  return make_float2(a.x * b.x - a.y * b.y, a.x * b.y + a.y * b.x);
}

// ---------------- in-place radix-2 complex FFT in LDS ----------------
// sgn = -1 forward, +1 inverse (unnormalized)
__device__ inline void cfft(float2* s, int n, int logn, float sgn) {
  const int tid = threadIdx.x, nt = blockDim.x;
  for (int i = tid; i < n; i += nt) {
    int j = (int)(__brev((unsigned)i) >> (32 - logn));
    if (j > i) { float2 t = s[i]; s[i] = s[j]; s[j] = t; }
  }
  __syncthreads();
  for (int st = 1; st <= logn; ++st) {
    const int half = 1 << (st - 1);
    const float w0 = sgn * PI_F / (float)half;
    for (int b = tid; b < (n >> 1); b += nt) {
      int j  = b & (half - 1);
      int i0 = ((b >> (st - 1)) << st) + j;
      int i1 = i0 + half;
      float sn, cs;
      __sincosf(w0 * (float)j, &sn, &cs);
      float2 u = s[i0], v = s[i1];
      float2 t = make_float2(v.x * cs - v.y * sn, v.x * sn + v.y * cs);
      s[i0] = make_float2(u.x + t.x, u.y + t.y);
      s[i1] = make_float2(u.x - t.x, u.y - t.y);
    }
    __syncthreads();
  }
}

// real-FFT unpack: X[k] from Z[k], Z[M-k]; th = pi*k/M
__device__ inline float2 runpack(float2 zk, float2 zmk, float th) {
  float ax = 0.5f * (zk.x + zmk.x), ay = 0.5f * (zk.y - zmk.y);
  float bx = 0.5f * (zk.x - zmk.x), by = 0.5f * (zk.y + zmk.y);
  float sn, cs; __sincosf(th, &sn, &cs);       // e^{-i th} = (cs, -sn)
  float tx = bx * cs + by * sn;
  float ty = by * cs - bx * sn;
  return make_float2(ax + ty, ay - tx);        // A + (-i)*(tx + i ty)
}

// inverse real-FFT pack: Z[k] from X[k], X[M-k]; th = pi*k/M
__device__ inline float2 rpack(float2 xk, float2 xmk, float th) {
  float ax = xk.x + xmk.x, ay = xk.y - xmk.y;
  float bx = xk.x - xmk.x, by = xk.y + xmk.y;
  float sn, cs; __sincosf(th, &sn, &cs);       // e^{+i th} = (cs, sn)
  float tx = bx * cs - by * sn;
  float ty = bx * sn + by * cs;
  return make_float2(0.5f * (ax - ty), 0.5f * (ay + tx));
}

// ---------------- kernel: per-atom scalar prep (1 block, 64 threads) -------
__global__ void k_prep(const float* __restrict__ mix, const float* __restrict__ decays,
                       const float* __restrict__ fdecays, const float* __restrict__ nfilt,
                       const float* __restrict__ rf1, const float* __restrict__ rf2,
                       const float* __restrict__ env, const float* __restrict__ amps,
                       const float* __restrict__ verb, const float* __restrict__ ln_w,
                       const float* __restrict__ ln_b, const float* __restrict__ mw1,
                       const float* __restrict__ mw2, const float* __restrict__ rw1,
                       const float* __restrict__ rw2, AtomParams* __restrict__ P) {
  int t = threadIdx.x;
  if (t >= N_ATOMS) return;
  float e0 = expf(mix[2 * t]), e1 = expf(mix[2 * t + 1]);
  P->m0[t] = e0 / (e0 + e1);
  P->m1[t] = e1 / (e0 + e1);
  float d = 0.02f + sigmoidf_(decays[t]) * 0.98f * 0.95f;
  P->ld_d[t] = logf(d + 1e-12f);
  float fd = 0.02f + sigmoidf_(fdecays[t]) * 0.98f * 0.95f;
  P->ld_f[t] = logf(fd + 1e-12f);
  P->mu_n[t] = nfilt[2 * t];
  P->sg_n[t] = fabsf(nfilt[2 * t + 1]) + 1e-12f;
  P->sg1[t]  = fabsf(rf1[2 * t + 1]) + 1e-12f;
  P->sg2[t]  = fabsf(rf2[2 * t + 1]) + 1e-12f;
  P->mu_e[t] = env[2 * t];
  P->sg_e[t] = fabsf(env[2 * t + 1] + 1e-12f) * 0.1f;
  P->amp[t]  = fabsf(amps[t]);
  // layernorm over 4 verb params
  float h[4], mn = 0.f, vr = 0.f;
  for (int i = 0; i < 4; ++i) mn += verb[4 * t + i];
  mn *= 0.25f;
  for (int i = 0; i < 4; ++i) { float dd = verb[4 * t + i] - mn; vr += dd * dd; }
  vr *= 0.25f;
  float inv = 1.f / sqrtf(vr + 1e-5f);
  for (int i = 0; i < 4; ++i) h[i] = (verb[4 * t + i] - mn) * inv * ln_w[i] + ln_b[i];
  // mix MLP -> sigmoid scalar
  float t1[4];
  for (int j = 0; j < 4; ++j) {
    float s = 0.f;
    for (int i = 0; i < 4; ++i) s += h[i] * mw1[4 * i + j];
    t1[j] = leakyf_(s);
  }
  float mv = 0.f;
  for (int j = 0; j < 4; ++j) mv += t1[j] * mw2[j];
  P->mixv[t] = sigmoidf_(mv);
  // room MLP -> argmax (forward value of sparse_softmax_st is the one-hot)
  float t2[4];
  for (int j = 0; j < 4; ++j) {
    float s = 0.f;
    for (int i = 0; i < 4; ++i) s += h[i] * rw1[4 * i + j];
    t2[j] = leakyf_(s);
  }
  int best = 0; float bv = -1e30f;
  for (int r = 0; r < 8; ++r) {
    float s = 0.f;
    for (int j = 0; j < 4; ++j) s += t2[j] * rw2[8 * j + r];
    if (s > bv) { bv = s; best = r; }
  }
  P->room[t] = best;
}

// ---------------- kernel: relu + f32->bf16 convert of A -------------------
__global__ void k_reluA(const float* __restrict__ rc, unsigned short* __restrict__ Abf) {
  int idx = blockIdx.x * 256 + threadIdx.x;   // 64*4096 total
  float v = rc[idx];
  v = v > 0.f ? v : 0.f;
  Abf[idx] = f2bfbits(v);
}

// ---------------- kernel: WMMA bf16 GEMM  R(64xNS) = A(64xK) * waves(KxNS) --
// Block = 8 wave32s. Block tile 64M x 64N, K-step 64, double-buffered LDS,
// one barrier per K-step. A tile is DMA'd with global_load_async_to_lds_b128
// (ASYNCcnt); B tile streams via VGPRs (needs f32->bf16 convert), issued one
// K-step ahead so VMEM latency hides under the WMMAs. Each thread stages a
// 4Kx4N sub-tile of B so converted bf16s pack into ds_store_b64.
// Wave w: mtile = w&3 (16 rows), nhalf = w>>2 (two 16-col tiles) -> 4 WMMA/step.
__global__ void __launch_bounds__(256) k_gemm(const unsigned short* __restrict__ Abf,
                                              const float* __restrict__ waves,
                                              float* __restrict__ R) {
  __shared__ unsigned short sA[2][64][72];   // [buf][m][k] stride 144B (16B aligned)
  __shared__ unsigned short sBt[2][64][72];  // [buf][n][k] transposed

  const int t = threadIdx.x;
  const int lane = t & 31, wave = t >> 5;
  const int mtile = wave & 3, nhalf = wave >> 2;
  const int n0 = blockIdx.x * 64;
  const int hi = lane >> 4, l15 = lane & 15;
  const int NT = KDIM / 64;

  v8f acc0 = {0.f, 0.f, 0.f, 0.f, 0.f, 0.f, 0.f, 0.f};
  v8f acc1 = {0.f, 0.f, 0.f, 0.f, 0.f, 0.f, 0.f, 0.f};

  // B staging mapping: each thread owns k rows 4*kq..+3, n cols nb..nb+3
  const int kq = t >> 4;           // 0..15 -> k base 4*kq
  const int nb = (t & 15) * 4;     // n base within tile

  // --- stage tile 0 ---
  {
    int c = t * 2;                 // A: 512 x 16B chunks, 2 per thread (DMA)
    for (int q = 0; q < 2; ++q, ++c) {
      int r = c >> 3, seg = (c & 7) * 8;
      async_load_b128((unsigned)(size_t)&sA[0][r][seg],
                      Abf + (size_t)r * KDIM + seg);
    }
    float4 bv[4];
    for (int j = 0; j < 4; ++j)
      bv[j] = *(const float4*)(waves + (size_t)(4 * kq + j) * NS + n0 + nb);
    const float* bf = (const float*)bv;      // bf[j*4 + e] = B[4kq+j][nb+e]
    for (int e = 0; e < 4; ++e) {
      uint2 d = make_uint2(pack2bf(bf[e], bf[4 + e]),
                           pack2bf(bf[8 + e], bf[12 + e]));
      *(uint2*)&sBt[0][nb + e][4 * kq] = d;  // one ds_store_b64 per column
    }
    wait_asynccnt0();
  }
  __syncthreads();

  const int m   = mtile * 16 + l15;
  const int nl0 = nhalf * 32 + l15;      // first 16-col tile's n index
  float4 pv[4];

  for (int it = 0; it < NT; ++it) {
    const int cur = it & 1, nxt = cur ^ 1;
    const int k0n = (it + 1) * 64;
    const bool more = (it + 1) < NT;

    if (more) {
      // A tile for next step: pure LDS DMA (no VGPRs touched)
      int c = t * 2;
      for (int q = 0; q < 2; ++q, ++c) {
        int r = c >> 3, seg = (c & 7) * 8;
        async_load_b128((unsigned)(size_t)&sA[nxt][r][seg],
                        Abf + (size_t)r * KDIM + k0n + seg);
      }
      // B tile for next step: global -> VGPR (consumed after the WMMAs)
      for (int j = 0; j < 4; ++j)
        pv[j] = *(const float4*)(waves + (size_t)(k0n + 4 * kq + j) * NS + n0 + nb);
      // prefetch B two steps ahead (global_prefetch_b8)
      if (it + 2 < NT)
        __builtin_prefetch(waves + (size_t)(k0n + 64 + 4 * kq) * NS + n0 + nb, 0, 1);
    }

    // --- compute on current tile: 2 K-chunks x 2 N-tiles = 4 WMMAs ---
#pragma unroll
    for (int kc = 0; kc < 2; ++kc) {
      const int kb = kc * 32;
      union { v16bf v; unsigned u[8]; } fa, fb0, fb1;
#pragma unroll
      for (int vv = 0; vv < 4; ++vv) {
        fa.u[vv]     = *(const unsigned*)&sA[cur][m][kb + hi * 16 + 2 * vv];
        fa.u[vv + 4] = *(const unsigned*)&sA[cur][m][kb + hi * 16 + 8 + 2 * vv];
      }
#pragma unroll
      for (int vv = 0; vv < 8; ++vv) {
        fb0.u[vv] = *(const unsigned*)&sBt[cur][nl0][kb + hi * 16 + 2 * vv];
        fb1.u[vv] = *(const unsigned*)&sBt[cur][nl0 + 16][kb + hi * 16 + 2 * vv];
      }
      acc0 = __builtin_amdgcn_wmma_f32_16x16x32_bf16(false, fa.v, false, fb0.v,
                                                     (short)0, acc0, false, false);
      acc1 = __builtin_amdgcn_wmma_f32_16x16x32_bf16(false, fa.v, false, fb1.v,
                                                     (short)0, acc1, false, false);
    }

    if (more) {
      // convert staged B VGPRs, pack pairs, ds_store_b64 into next buffer
      const float* bf = (const float*)pv;
      for (int e = 0; e < 4; ++e) {
        uint2 d = make_uint2(pack2bf(bf[e], bf[4 + e]),
                             pack2bf(bf[8 + e], bf[12 + e]));
        *(uint2*)&sBt[nxt][nb + e][4 * kq] = d;
      }
    }
    wait_asynccnt0();   // A DMA into next buffer complete
    __syncthreads();    // single barrier per K-step (double-buffered)
  }

  const int colbase = n0 + nhalf * 32 + l15;
#pragma unroll
  for (int r = 0; r < 8; ++r) {
    const int row = mtile * 16 + hi * 8 + r;   // C/D layout: lanes16-31 hold M+8
    R[(size_t)row * NS + colbase]      = acc0[r];
    R[(size_t)row * NS + colbase + 16] = acc1[r];
  }
}

// ---------------- kernel: spectral gaussian filter of a 32768-pt real row ---
// dst_row = irfft(rfft(src_row) * gauss(mu, sg)); one workgroup per atom.
__global__ void k_gauss(const float* __restrict__ src, float* __restrict__ dst,
                        const AtomParams* __restrict__ P, int mode) {
  extern __shared__ float2 sh[];
  const int a = blockIdx.x, tid = threadIdx.x, nt = blockDim.x;
  float mu, sg;
  if (mode == 0)      { mu = P->mu_n[a]; sg = P->sg_n[a]; }
  else if (mode == 1) { mu = 0.f;        sg = P->sg1[a];  }
  else                { mu = 0.f;        sg = P->sg2[a];  }
  const float* row = src + (size_t)a * NS;
  for (int k = tid; k < MFILT; k += nt)
    sh[k] = make_float2(row[2 * k], row[2 * k + 1]);
  __syncthreads();
  cfft(sh, MFILT, LOG_MFILT, -1.f);
  const float c = PI_F / (float)MFILT;
  const float fstep = 1.f / (float)(NCOEF - 1);
  for (int k = tid; k <= MFILT / 2; k += nt) {
    int km = MFILT - k;
    float2 zk  = sh[k & (MFILT - 1)];
    float2 zmk = sh[km & (MFILT - 1)];
    float2 xk  = runpack(zk, zmk, c * (float)k);
    float2 xmk = runpack(zmk, zk, c * (float)km);
    float gk = gaussf_((float)k * fstep, mu, sg);
    float gm = gaussf_((float)km * fstep, mu, sg);
    xk  = make_float2(xk.x * gk, xk.y * gk);
    xmk = make_float2(xmk.x * gm, xmk.y * gm);
    float2 nzk = rpack(xk, xmk, c * (float)k);
    sh[k & (MFILT - 1)] = nzk;                       // k==0 writes only Z'[0]
    if (k != 0 && k != MFILT / 2)
      sh[km] = rpack(xmk, xk, c * (float)km);
  }
  __syncthreads();
  cfft(sh, MFILT, LOG_MFILT, +1.f);
  float* out = dst + (size_t)a * NS;
  const float s = 1.f / (float)MFILT;
  for (int k = tid; k < MFILT; k += nt) {
    float2 z = sh[k];
    out[2 * k]     = z.x * s;
    out[2 * k + 1] = z.y * s;
  }
}

// ---------------- kernel: apply gaussian time envelope in place ------------
__global__ void k_env(float* __restrict__ buf, const AtomParams* __restrict__ P) {
  int idx = blockIdx.x * 256 + threadIdx.x;
  int a = idx >> 15, i = idx & (NS - 1);
  float rng = (float)i * (1.f / (float)(NS - 1));
  buf[idx] *= gaussf_(rng, P->mu_e[a], P->sg_e[a]);
}

// ---------------- kernel: 65536-pt rfft of zero-padded 32768 row -----------
// mode 0: plain; mode 1: multiply by decay envelope; mode 2: gather room IR.
__global__ void k_rfft(const float* __restrict__ src, float2* __restrict__ spec,
                       const AtomParams* __restrict__ P, int mode) {
  extern __shared__ float2 sh[];
  const int a = blockIdx.x, tid = threadIdx.x, nt = blockDim.x;
  const float* row = (mode == 2) ? (src + (size_t)P->room[a] * NS)
                                 : (src + (size_t)a * NS);
  const float ld = P->ld_d[a];
  for (int k = tid; k < MCONV; k += nt) {
    float x0 = 0.f, x1 = 0.f;
    int i = 2 * k;
    if (i < NS) {                       // upper half of padded signal is zero
      x0 = row[i];
      x1 = row[i + 1];
      if (mode == 1) {
        x0 *= decay_interp(ld, i);
        x1 *= decay_interp(ld, i + 1);
      }
    }
    sh[k] = make_float2(x0, x1);
  }
  __syncthreads();
  cfft(sh, MCONV, LOG_MCONV, -1.f);
  float2* out = spec + (size_t)a * (MCONV + 1);
  const float c = PI_F / (float)MCONV;
  for (int k = tid; k <= MCONV; k += nt) {
    float2 zk  = sh[k & (MCONV - 1)];
    float2 zmk = sh[(MCONV - k) & (MCONV - 1)];
    out[k] = runpack(zk, zmk, c * (float)k);
  }
}

// ---------------- kernel: spectrum product + 65536-pt irfft (keep first NS) -
__global__ void k_specmul(const float2* __restrict__ sa, const float2* __restrict__ sb,
                          float* __restrict__ dst) {
  extern __shared__ float2 sh[];
  const int a = blockIdx.x, tid = threadIdx.x, nt = blockDim.x;
  const float2* ra = sa + (size_t)a * (MCONV + 1);
  const float2* rb = sb + (size_t)a * (MCONV + 1);
  const float c = PI_F / (float)MCONV;
  for (int k = tid; k < MCONV; k += nt) {
    float2 xk  = cmul_(ra[k], rb[k]);
    float2 xmk = cmul_(ra[MCONV - k], rb[MCONV - k]);
    sh[k] = rpack(xk, xmk, c * (float)k);
  }
  __syncthreads();
  cfft(sh, MCONV, LOG_MCONV, +1.f);
  float* out = dst + (size_t)a * NS;
  const float s = 1.f / (float)MCONV;
  for (int k = tid; k < NS / 2; k += nt) {
    float2 z = sh[k];
    out[2 * k]     = z.x * s;
    out[2 * k + 1] = z.y * s;
  }
}

// ---------------- kernel: crossfade, mix, L2-normalize, amp scale ----------
__global__ void k_final(const float* __restrict__ Pn, const float* __restrict__ r1,
                        const float* __restrict__ r2, float* __restrict__ F,
                        float* __restrict__ outbuf, const AtomParams* __restrict__ P) {
  __shared__ float red[256];
  const int a = blockIdx.x, tid = threadIdx.x;
  const size_t base = (size_t)a * NS;
  const float m0 = P->m0[a], m1 = P->m1[a], ldf = P->ld_f[a];
  float ss = 0.f;
  for (int i = tid; i < NS; i += 256) {
    float cf = decay_interp(ldf, i);
    float mx = cf * r1[base + i] + (1.f - cf) * r2[base + i];
    float f  = Pn[base + i] * m0 + mx * m1;
    F[base + i] = f;
    ss += f * f;
  }
  red[tid] = ss;
  __syncthreads();
  for (int s = 128; s > 0; s >>= 1) {
    if (tid < s) red[tid] += red[tid + s];
    __syncthreads();
  }
  const float scale = P->amp[a] / (sqrtf(red[0]) + 1e-8f);
  for (int i = tid; i < NS; i += 256) F[base + i] *= scale;
  if (tid == 0) outbuf[TOTAL + a] = P->amp[a];   // second tuple output: amps
}

// ---------------- kernel: dry/wet blend to d_out ---------------------------
__global__ void k_out(const float* __restrict__ F, const float* __restrict__ wet,
                      const AtomParams* __restrict__ P, float* __restrict__ out) {
  int idx = blockIdx.x * 256 + threadIdx.x;
  int a = idx >> 15;
  float mv = P->mixv[a];
  out[idx] = F[idx] * mv + wet[idx] * (1.f - mv);
}

// ---------------- launcher -------------------------------------------------
extern "C" void kernel_launch(void* const* d_in, const int* in_sizes, int n_in,
                              void* d_out, int out_size, void* d_ws, size_t ws_size,
                              hipStream_t stream) {
  const float* mix      = (const float*)d_in[1];
  const float* decays   = (const float*)d_in[2];
  const float* fdecays  = (const float*)d_in[3];
  const float* rchoice  = (const float*)d_in[4];
  const float* nfilt    = (const float*)d_in[5];
  const float* rf1      = (const float*)d_in[6];
  const float* rf2      = (const float*)d_in[7];
  const float* env      = (const float*)d_in[8];
  const float* amps     = (const float*)d_in[9];
  const float* verb     = (const float*)d_in[10];
  const float* noise    = (const float*)d_in[11];
  const float* waves    = (const float*)d_in[12];
  const float* ln_w     = (const float*)d_in[13];
  const float* ln_b     = (const float*)d_in[14];
  const float* mw1      = (const float*)d_in[15];
  const float* mw2      = (const float*)d_in[16];
  const float* rw1      = (const float*)d_in[17];
  const float* rw2      = (const float*)d_in[18];
  const float* room_irs = (const float*)d_in[19];
  float* out = (float*)d_out;

  // workspace layout (floats)
  float* ws = (float*)d_ws;
  AtomParams* P = (AtomParams*)ws;                       // ~3.4 KB
  unsigned short* Abf = (unsigned short*)(ws + 4096);    // 64*4096 bf16
  float* B0 = ws + 4096 + 131072;                        // resonances
  float* B1 = B0 + (size_t)TOTAL;                        // filt noise -> positioned -> wet
  float* B2 = B1 + (size_t)TOTAL;                        // filt res1  -> res
  float* B3 = B2 + (size_t)TOTAL;                        // filt res2  -> res2
  float* B4 = B3 + (size_t)TOTAL;                        // final
  float2* SP = (float2*)(B4 + (size_t)TOTAL);            // spectra: 64 x 32769 c64
  float2* S1 = SP + (size_t)N_ATOMS * (MCONV + 1);
  float2* S2 = S1 + (size_t)N_ATOMS * (MCONV + 1);

  const size_t LDS_CONV = (size_t)MCONV * sizeof(float2); // 256 KB (fits 320KB WGP)
  const size_t LDS_FILT = (size_t)MFILT * sizeof(float2); // 128 KB

  k_prep<<<1, 64, 0, stream>>>(mix, decays, fdecays, nfilt, rf1, rf2, env, amps,
                               verb, ln_w, ln_b, mw1, mw2, rw1, rw2, P);
  k_reluA<<<(N_ATOMS * KDIM) / 256, 256, 0, stream>>>(rchoice, Abf);
  k_gemm<<<NS / 64, 256, 0, stream>>>(Abf, waves, B0);

  k_gauss<<<N_ATOMS, 1024, LDS_FILT, stream>>>(noise, B1, P, 0);  // filtered noise
  k_gauss<<<N_ATOMS, 1024, LDS_FILT, stream>>>(B0,    B2, P, 1);  // filtered res (rf1)
  k_gauss<<<N_ATOMS, 1024, LDS_FILT, stream>>>(B0,    B3, P, 2);  // filtered res (rf2)

  k_env<<<TOTAL / 256, 256, 0, stream>>>(B1, P);                  // positioned noise

  k_rfft<<<N_ATOMS, 1024, LDS_CONV, stream>>>(B1, SP, P, 0);      // spec(P)
  k_rfft<<<N_ATOMS, 1024, LDS_CONV, stream>>>(B2, S1, P, 1);      // spec(res1*decay)
  k_rfft<<<N_ATOMS, 1024, LDS_CONV, stream>>>(B3, S2, P, 1);      // spec(res2*decay)
  k_specmul<<<N_ATOMS, 1024, LDS_CONV, stream>>>(SP, S1, B2);     // res
  k_specmul<<<N_ATOMS, 1024, LDS_CONV, stream>>>(SP, S2, B3);     // res2

  k_final<<<N_ATOMS, 256, 0, stream>>>(B1, B2, B3, B4, out, P);   // final + amps

  k_rfft<<<N_ATOMS, 1024, LDS_CONV, stream>>>(B4, S1, P, 0);      // spec(final)
  k_rfft<<<N_ATOMS, 1024, LDS_CONV, stream>>>(room_irs, S2, P, 2);// spec(room IR)
  k_specmul<<<N_ATOMS, 1024, LDS_CONV, stream>>>(S1, S2, B1);     // wet

  k_out<<<TOTAL / 256, 256, 0, stream>>>(B4, B1, P, out);
  (void)in_sizes; (void)n_in; (void)out_size; (void)ws_size;
}
